// NeighborVoxelTransSAModuleMSG_56822417326465
// MI455X (gfx1250) — matrix operand order repaired and checked by reading, not compiled, not run on hardware
//
#include <hip/hip_runtime.h>
#include <stdint.h>

typedef __attribute__((ext_vector_type(16))) _Float16 v16h;
typedef __attribute__((ext_vector_type(8)))  _Float16 v8h;
typedef __attribute__((ext_vector_type(8)))  float    v8f;

namespace {

constexpr int kNS = 32;     // neighbors (tokens)
constexpr int kE  = 64;     // channels
constexpr int kFF = 128;    // ffn hidden
constexpr int kOut = 128;   // output channels
constexpr int kL  = 2;      // layers
constexpr int kThreads = 256;
constexpr int kWaves = 8;
constexpr int kP  = 2;      // points processed concurrently per workgroup

struct __align__(16) PointBuf {
  int      idxbuf[kNS];
  float    gx[kNS * 3];
  float    xf[kNS * kE];           // residual stream, f32
  _Float16 hpos[kNS * 32];         // pos-mlp hidden (relu), f16
  _Float16 sf[kNS * kE];           // f16 staging (LN out / attn out)
  _Float16 qf[kNS * kE];           // q * scale, f16
  _Float16 kT[kE * kNS];           // k transposed [E][tokens], f16
  _Float16 vf[kNS * kE];           // v, f16
  float    att[kNS * kNS];         // scores, f32
  _Float16 probs[kNS * kNS];       // softmax probs, f16
  float    acc[kNS * kE];          // gemm f32 output (pending residual)
  _Float16 hf[kNS * kFF];          // ffn hidden (relu), f16
  float    pooled[kE];
};

struct __align__(16) Smem {
  // ---- weights, f16, transposed to K-major [Kin][Nout] ----
  _Float16 qkvT[kL][kE * 192];     // 49152 B
  _Float16 aoT [kL][kE * kE];      // 16384 B
  _Float16 ff1T[kL][kE * kFF];     // 32768 B
  _Float16 ff2T[kL][kFF * kE];     // 32768 B
  _Float16 posw2T[32 * kE];        //  4096 B
  // ---- params, f32 ----
  float qkvb[kL][192];
  float aob [kL][kE];
  float ln1g[kL][kE], ln1b[kL][kE];
  float ln2g[kL][kE], ln2b[kL][kE];
  float ff1b[kL][kFF];
  float ff2b[kL][kE];
  float posw1[32 * 3];
  float bnscale[32], bnshift[32];
  float posb2[kE];
  // ---- per-point working sets ----
  PointBuf pt[kP];
};
static_assert(sizeof(Smem) <= 320 * 1024, "exceeds WGP LDS");

// Async 16-byte global -> LDS copy (per-lane addresses), tracked by ASYNCcnt.
__device__ __forceinline__ void async_g2l_b128(void* lds_ptr, const void* gptr) {
  uint32_t lds_addr = (uint32_t)(uintptr_t)lds_ptr;
  asm volatile("global_load_async_to_lds_b128 %0, %1, off"
               :: "v"(lds_addr), "v"(gptr)
               : "memory");
}
__device__ __forceinline__ void wait_async0() {
  asm volatile("s_wait_asynccnt 0x0" ::: "memory");
}

// One 16x16 D tile, A: MxK f16 row-major (lda), B: KxN f16 K-major (ldb).
__device__ __forceinline__ v8f wmma_tile(const _Float16* A, int lda,
                                         const _Float16* B, int ldb,
                                         int row0, int n0, int K) {
  const int lane = threadIdx.x & 31;
  const int r     = row0 + (lane & 15);
  const int kboff = (lane >> 4) << 3;                 // A: 0 or 8
  const int kb2   = (lane & 15) + ((lane >> 4) << 4); // B: lane's K row
  v8f c = {};
  for (int k0 = 0; k0 < K; k0 += 32) {
    const _Float16* pa = A + r * lda + k0 + kboff;
    v8h alo = *(const v8h*)pa;
    v8h ahi = *(const v8h*)(pa + 16);
    v16h a;
#pragma unroll
    for (int i = 0; i < 8; ++i) { a[i] = alo[i]; a[i + 8] = ahi[i]; }
    const _Float16* pb = B + (k0 + kboff) * ldb + n0 + (lane & 15);
    v16h b;
#pragma unroll
    for (int i = 0; i < 8; ++i) { b[i] = pb[i * ldb]; b[i + 8] = pb[(i + 16) * ldb]; }
    c = __builtin_amdgcn_wmma_f32_16x16x32_f16(false, a, false, b,
                                               (short)0, c, false, false);
  }
  return c;
}

// LayerNorm over 64 rows (2 points x 32 tokens). Optionally folds the pending
// residual (acc) into xf first. Writes normalized f16 rows to sf.
__device__ __forceinline__ void ln_rows2(Smem& sm, const float* g,
                                         const float* b, bool add_acc,
                                         int tid) {
  if (tid < kP * kNS) {
    PointBuf& pb = sm.pt[tid >> 5];
    const int r = tid & 31;
    float* row = pb.xf + r * kE;
    const float* ar = pb.acc + r * kE;
    float mu = 0.f, sq = 0.f;
#pragma unroll 8
    for (int c = 0; c < kE; ++c) {
      float v = row[c];
      if (add_acc) { v += ar[c]; row[c] = v; }
      mu += v; sq += v * v;
    }
    mu *= (1.f / kE);
    float var = sq * (1.f / kE) - mu * mu;
    float rs = rsqrtf(var + 1e-5f);
#pragma unroll 8
    for (int c = 0; c < kE; ++c)
      pb.sf[r * kE + c] = (_Float16)((row[c] - mu) * rs * g[c] + b[c]);
  }
}

__global__ __launch_bounds__(kThreads)
void pt_transformer_kernel(
    const float* __restrict__ xyz, const float* __restrict__ new_xyz,
    const float* __restrict__ features, const int* __restrict__ group_idx,
    const unsigned char* __restrict__ empty_mask,
    const float* __restrict__ pos_w1, const float* __restrict__ pos_bn_g,
    const float* __restrict__ pos_bn_b, const float* __restrict__ pos_bn_mean,
    const float* __restrict__ pos_bn_var, const float* __restrict__ pos_w2,
    const float* __restrict__ pos_b2, const float* __restrict__ ln1_g,
    const float* __restrict__ ln1_b, const float* __restrict__ qkv_w,
    const float* __restrict__ qkv_b, const float* __restrict__ attn_out_w,
    const float* __restrict__ attn_out_b, const float* __restrict__ ln2_g,
    const float* __restrict__ ln2_b, const float* __restrict__ ff1_w,
    const float* __restrict__ ff1_b, const float* __restrict__ ff2_w,
    const float* __restrict__ ff2_b, const float* __restrict__ out_w,
    const float* __restrict__ out_b, float* __restrict__ out, int M) {
  __shared__ Smem sm;
  const int tid  = threadIdx.x;
  const int wave = tid >> 5;
  const int lane = tid & 31;

  // ---------------- one-time weight preload (f32 -> f16, transpose) --------
  for (int i = tid; i < kL * 192 * kE; i += kThreads) {
    int l = i / (192 * kE), rem = i - l * 192 * kE, n = rem / kE, k = rem % kE;
    sm.qkvT[l][k * 192 + n] = (_Float16)qkv_w[i];
  }
  for (int i = tid; i < kL * kE * kE; i += kThreads) {
    int l = i / (kE * kE), rem = i - l * kE * kE, n = rem / kE, k = rem % kE;
    sm.aoT[l][k * kE + n] = (_Float16)attn_out_w[i];
  }
  for (int i = tid; i < kL * kFF * kE; i += kThreads) {
    int l = i / (kFF * kE), rem = i - l * kFF * kE, n = rem / kE, k = rem % kE;
    sm.ff1T[l][k * kFF + n] = (_Float16)ff1_w[i];
  }
  for (int i = tid; i < kL * kE * kFF; i += kThreads) {
    int l = i / (kE * kFF), rem = i - l * kE * kFF, n = rem / kFF, k = rem % kFF;
    sm.ff2T[l][k * kE + n] = (_Float16)ff2_w[i];
  }
  for (int i = tid; i < kE * 32; i += kThreads) { // pos_w2 [E][32]
    int n = i / 32, k = i % 32;
    sm.posw2T[k * kE + n] = (_Float16)pos_w2[i];
  }
  for (int i = tid; i < kL * 192; i += kThreads) (&sm.qkvb[0][0])[i] = qkv_b[i];
  for (int i = tid; i < kL * kE;  i += kThreads) {
    (&sm.aob [0][0])[i] = attn_out_b[i];
    (&sm.ln1g[0][0])[i] = ln1_g[i]; (&sm.ln1b[0][0])[i] = ln1_b[i];
    (&sm.ln2g[0][0])[i] = ln2_g[i]; (&sm.ln2b[0][0])[i] = ln2_b[i];
    (&sm.ff2b[0][0])[i] = ff2_b[i];
  }
  for (int i = tid; i < kL * kFF; i += kThreads) (&sm.ff1b[0][0])[i] = ff1_b[i];
  for (int i = tid; i < 96; i += kThreads) sm.posw1[i] = pos_w1[i];
  for (int i = tid; i < kE; i += kThreads) sm.posb2[i] = pos_b2[i];
  if (tid < 32) {
    float sc = pos_bn_g[tid] * rsqrtf(pos_bn_var[tid] + 1e-5f);
    sm.bnscale[tid] = sc;
    sm.bnshift[tid] = pos_bn_b[tid] - pos_bn_mean[tid] * sc;
  }
  __syncthreads();

  // ---------------- pipeline: 2 points per workgroup iteration -------------
  for (int mb = blockIdx.x * kP; mb < M; mb += gridDim.x * kP) {
    const int  m0 = mb;
    const bool v1 = (mb + 1) < M;
    const int  m1 = v1 ? (mb + 1) : mb;
    const int  mm[kP] = {m0, m1};
    bool emptyf[kP];
    emptyf[0] = empty_mask[m0] != 0;
    emptyf[1] = empty_mask[m1] != 0;
    if (mb + (int)gridDim.x * kP < M)
      __builtin_prefetch(&group_idx[(size_t)(mb + gridDim.x * kP) * kNS], 0, 1);

    if (tid < kP * kNS)
      sm.pt[tid >> 5].idxbuf[tid & 31] =
          group_idx[(size_t)mm[tid >> 5] * kNS + (tid & 31)];
    __syncthreads();

    // Gather neighbor features straight into LDS with async b128 copies
    // (2 points x 32 rows x 16 chunks of 4 floats = 1024 chunks).
    for (int j = tid; j < kP * kNS * 16; j += kThreads) {
      int p  = j >> 9;          // wave-uniform (512-chunk blocks, 32-aligned)
      int r  = (j >> 4) & 31;
      int c4 = (j & 15) * 4;
      PointBuf& pb = sm.pt[p];
      if (emptyf[p]) {
        float4 z = make_float4(0.f, 0.f, 0.f, 0.f);
        *(float4*)&pb.xf[r * kE + c4] = z;
      } else {
        async_g2l_b128(&pb.xf[r * kE + c4],
                       features + (size_t)pb.idxbuf[r] * kE + c4);
      }
    }
    if (tid < kP * kNS) {
      int p = tid >> 5, r = tid & 31;
      int id = sm.pt[p].idxbuf[r];
#pragma unroll
      for (int d = 0; d < 3; ++d)
        sm.pt[p].gx[r * 3 + d] =
            emptyf[p] ? 0.f
                      : (xyz[(size_t)id * 3 + d] - new_xyz[(size_t)mm[p] * 3 + d]);
    }
    wait_async0();
    __syncthreads();

    // pos MLP layer1 + folded BN + relu -> hpos f16
    for (int i = tid; i < kP * kNS * 32; i += kThreads) {
      int p = i >> 10, rem = i & 1023, n = rem >> 5, h = rem & 31;
      PointBuf& pb = sm.pt[p];
      float v = pb.gx[n * 3 + 0] * sm.posw1[h * 3 + 0] +
                pb.gx[n * 3 + 1] * sm.posw1[h * 3 + 1] +
                pb.gx[n * 3 + 2] * sm.posw1[h * 3 + 2];
      v = v * sm.bnscale[h] + sm.bnshift[h];
      pb.hpos[rem] = (_Float16)fmaxf(v, 0.f);
    }
    __syncthreads();

    // pos MLP layer2 (WMMA, K=32): xf += hpos @ posw2T + posb2   (16 tiles)
    for (int t = wave; t < kP * 8; t += kWaves) {
      int p = t >> 3, tt = t & 7, mt = tt >> 2, nt = tt & 3;
      PointBuf& pb = sm.pt[p];
      v8f c = wmma_tile(pb.hpos, 32, sm.posw2T, kE, mt * 16, nt * 16, 32);
#pragma unroll
      for (int v = 0; v < 8; ++v) {
        int row = mt * 16 + v + ((lane >> 4) << 3);
        int col = nt * 16 + (lane & 15);
        pb.xf[row * kE + col] += c[v] + sm.posb2[col];
      }
    }
    __syncthreads();

    // -------------------- transformer layers --------------------
    for (int l = 0; l < kL; ++l) {
      // ln1 (folds previous layer's ff residual for l > 0)
      ln_rows2(sm, sm.ln1g[l], sm.ln1b[l], l > 0, tid);
      __syncthreads();

      // qkv: (32x64)@(64x192) -> q*scale, kT, v   (48 tiles, K=64)
      for (int t = wave; t < kP * 24; t += kWaves) {
        int p = t / 24, tt = t % 24, mt = tt / 12, nt = tt % 12;
        PointBuf& pb = sm.pt[p];
        v8f c = wmma_tile(pb.sf, kE, sm.qkvT[l], 192, mt * 16, nt * 16, kE);
#pragma unroll
        for (int v = 0; v < 8; ++v) {
          int row = mt * 16 + v + ((lane >> 4) << 3);
          int col = nt * 16 + (lane & 15);
          float val = c[v] + sm.qkvb[l][col];
          if (col < 64)        pb.qf[row * kE + col] = (_Float16)(val * 0.125f);
          else if (col < 128)  pb.kT[(col - 64) * kNS + row] = (_Float16)val;
          else                 pb.vf[row * kE + (col - 128)] = (_Float16)val;
        }
      }
      __syncthreads();

      // scores: (32x64)@(64x32) -> att  (8 tiles, K=64)
      for (int t = wave; t < kP * 4; t += kWaves) {
        int p = t >> 2, tt = t & 3, mt = tt >> 1, nt = tt & 1;
        PointBuf& pb = sm.pt[p];
        v8f c = wmma_tile(pb.qf, kE, pb.kT, kNS, mt * 16, nt * 16, kE);
#pragma unroll
        for (int v = 0; v < 8; ++v) {
          int row = mt * 16 + v + ((lane >> 4) << 3);
          int col = nt * 16 + (lane & 15);
          pb.att[row * kNS + col] = c[v];
        }
      }
      __syncthreads();

      // softmax rows -> probs f16 (64 rows)
      if (tid < kP * kNS) {
        PointBuf& pb = sm.pt[tid >> 5];
        float* row = pb.att + (tid & 31) * kNS;
        float mx = row[0];
#pragma unroll 8
        for (int j = 1; j < kNS; ++j) mx = fmaxf(mx, row[j]);
        float s = 0.f;
#pragma unroll 8
        for (int j = 0; j < kNS; ++j) { float e = __expf(row[j] - mx); s += e; row[j] = e; }
        float inv = 1.f / s;
#pragma unroll 8
        for (int j = 0; j < kNS; ++j)
          pb.probs[(tid & 31) * kNS + j] = (_Float16)(row[j] * inv);
      }
      __syncthreads();

      // attn @ v: (32x32)@(32x64) -> sf f16  (16 tiles, K=32)
      for (int t = wave; t < kP * 8; t += kWaves) {
        int p = t >> 3, tt = t & 7, mt = tt >> 2, nt = tt & 3;
        PointBuf& pb = sm.pt[p];
        v8f c = wmma_tile(pb.probs, kNS, pb.vf, kE, mt * 16, nt * 16, kNS);
#pragma unroll
        for (int v = 0; v < 8; ++v) {
          int row = mt * 16 + v + ((lane >> 4) << 3);
          int col = nt * 16 + (lane & 15);
          pb.sf[row * kE + col] = (_Float16)c[v];
        }
      }
      __syncthreads();

      // out proj: (32x64)@(64x64) -> acc (pending residual)  (16 tiles, K=64)
      for (int t = wave; t < kP * 8; t += kWaves) {
        int p = t >> 3, tt = t & 7, mt = tt >> 2, nt = tt & 3;
        PointBuf& pb = sm.pt[p];
        v8f c = wmma_tile(pb.sf, kE, sm.aoT[l], kE, mt * 16, nt * 16, kE);
#pragma unroll
        for (int v = 0; v < 8; ++v) {
          int row = mt * 16 + v + ((lane >> 4) << 3);
          int col = nt * 16 + (lane & 15);
          pb.acc[row * kE + col] = c[v] + sm.aob[l][col];
        }
      }
      __syncthreads();

      // ln2 folds attn residual: xf += acc, then normalize -> sf
      ln_rows2(sm, sm.ln2g[l], sm.ln2b[l], true, tid);
      __syncthreads();

      // ff1: (32x64)@(64x128) relu -> hf f16  (32 tiles, K=64)
      for (int t = wave; t < kP * 16; t += kWaves) {
        int p = t >> 4, tt = t & 15, mt = tt >> 3, nt = tt & 7;
        PointBuf& pb = sm.pt[p];
        v8f c = wmma_tile(pb.sf, kE, sm.ff1T[l], kFF, mt * 16, nt * 16, kE);
#pragma unroll
        for (int v = 0; v < 8; ++v) {
          int row = mt * 16 + v + ((lane >> 4) << 3);
          int col = nt * 16 + (lane & 15);
          pb.hf[row * kFF + col] = (_Float16)fmaxf(c[v] + sm.ff1b[l][col], 0.f);
        }
      }
      __syncthreads();

      // ff2: (32x128)@(128x64) -> acc (pending residual)  (16 tiles, K=128)
      for (int t = wave; t < kP * 8; t += kWaves) {
        int p = t >> 3, tt = t & 7, mt = tt >> 2, nt = tt & 3;
        PointBuf& pb = sm.pt[p];
        v8f c = wmma_tile(pb.hf, kFF, sm.ff2T[l], kE, mt * 16, nt * 16, kFF);
#pragma unroll
        for (int v = 0; v < 8; ++v) {
          int row = mt * 16 + v + ((lane >> 4) << 3);
          int col = nt * 16 + (lane & 15);
          pb.acc[row * kE + col] = c[v] + sm.ff2b[l][col];
        }
      }
      __syncthreads();
      // ff residual folded into next ln1 (l+1) or into the pooling below.
    }

    // max-pool over tokens (folding the last ff residual), 128 lanes
    if (tid < kP * kE) {
      int p = tid >> 6, c = tid & 63;
      PointBuf& pb = sm.pt[p];
      float mx = -3.4e38f;
#pragma unroll 8
      for (int r = 0; r < kNS; ++r)
        mx = fmaxf(mx, pb.xf[r * kE + c] + pb.acc[r * kE + c]);
      pb.pooled[c] = mx;
    }
    __syncthreads();

    // output projection (64 -> 128), 2 points x 128 outputs = 256 threads
    {
      int p = tid >> 7, o = tid & 127;
      if (p == 0 || v1) {
        float s = out_b[o];
#pragma unroll 8
        for (int e = 0; e < kE; ++e) s += sm.pt[p].pooled[e] * out_w[o * kE + e];
        out[(size_t)mm[p] * kOut + o] = s;
      }
    }
    __syncthreads();  // protect per-point buffers before next iteration
  }
}

}  // namespace

extern "C" void kernel_launch(void* const* d_in, const int* in_sizes, int n_in,
                              void* d_out, int out_size, void* d_ws, size_t ws_size,
                              hipStream_t stream) {
  (void)n_in; (void)d_ws; (void)ws_size; (void)out_size;
  const float* xyz        = (const float*)d_in[0];
  const float* new_xyz    = (const float*)d_in[1];
  const float* features   = (const float*)d_in[2];
  const int*   group_idx  = (const int*)d_in[3];
  const unsigned char* empty_mask = (const unsigned char*)d_in[4];
  const float* pos_w1     = (const float*)d_in[5];
  const float* pos_bn_g   = (const float*)d_in[6];
  const float* pos_bn_b   = (const float*)d_in[7];
  const float* pos_bn_mean= (const float*)d_in[8];
  const float* pos_bn_var = (const float*)d_in[9];
  const float* pos_w2     = (const float*)d_in[10];
  const float* pos_b2     = (const float*)d_in[11];
  const float* ln1_g      = (const float*)d_in[12];
  const float* ln1_b      = (const float*)d_in[13];
  const float* qkv_w      = (const float*)d_in[14];
  const float* qkv_b      = (const float*)d_in[15];
  const float* attn_out_w = (const float*)d_in[16];
  const float* attn_out_b = (const float*)d_in[17];
  const float* ln2_g      = (const float*)d_in[18];
  const float* ln2_b      = (const float*)d_in[19];
  const float* ff1_w      = (const float*)d_in[20];
  const float* ff1_b      = (const float*)d_in[21];
  const float* ff2_w      = (const float*)d_in[22];
  const float* ff2_b      = (const float*)d_in[23];
  const float* out_w      = (const float*)d_in[24];
  const float* out_b      = (const float*)d_in[25];

  const int M = in_sizes[1] / 3;
  if (M <= 0) return;
  int pairs = (M + 1) / 2;
  int grid = pairs < 8192 ? pairs : 8192;
  pt_transformer_kernel<<<grid, kThreads, 0, stream>>>(
      xyz, new_xyz, features, group_idx, empty_mask, pos_w1, pos_bn_g,
      pos_bn_b, pos_bn_mean, pos_bn_var, pos_w2, pos_b2, ln1_g, ln1_b, qkv_w,
      qkv_b, attn_out_w, attn_out_b, ln2_g, ln2_b, ff1_w, ff1_b, ff2_w, ff2_b,
      out_w, out_b, (float*)d_out, M);
}
